// DecoderBlock_49263274885312
// MI455X (gfx1250) — matrix operand order repaired and checked by reading, not compiled, not run on hardware
//
#include <hip/hip_runtime.h>
#include <hip/hip_bf16.h>

typedef _Float16 half_t;
typedef __attribute__((ext_vector_type(16))) _Float16 v16h;
typedef __attribute__((ext_vector_type(8)))  _Float16 v8h;
typedef __attribute__((ext_vector_type(8)))  float    v8f;

#define D_MODEL 768
#define S_LEN   1024
#define BATCH   8
#define NHEADS  4
#define DHEAD   192
#define ROWS    (BATCH*S_LEN)   // 8192

enum { EPI_QKV = 0, EPI_SCORE = 1, EPI_ATTNOUT = 2, EPI_BIASRES = 3, EPI_GELU = 4 };

__device__ __forceinline__ float gelu_t(float x) {
    float x3 = x * x * x;
    return 0.5f * x * (1.0f + tanhf(0.7978845608028654f * (x + 0.044715f * x3)));
}

// ---------------------------------------------------------------------------
// fp32 -> f16 elementwise convert (weights)
// ---------------------------------------------------------------------------
__global__ __launch_bounds__(256) void f32_to_f16_k(const float* __restrict__ in,
                                                    half_t* __restrict__ out, int n) {
    int i = blockIdx.x * 256 + threadIdx.x;
    if (i < n) out[i] = (half_t)in[i];
}

// ---------------------------------------------------------------------------
// LayerNorm: one 256-thread block per row of 768, writes f16
// ---------------------------------------------------------------------------
__global__ __launch_bounds__(256) void layernorm_k(const float* __restrict__ x,
                                                   const float* __restrict__ g,
                                                   const float* __restrict__ b,
                                                   half_t* __restrict__ out) {
    const int row = blockIdx.x;
    const float* xr = x + (long long)row * D_MODEL;
    float v[3];
    float lsum = 0.f, lsq = 0.f;
#pragma unroll
    for (int i = 0; i < 3; ++i) {
        v[i] = xr[threadIdx.x + i * 256];
        lsum += v[i];
        lsq  += v[i] * v[i];
    }
    __shared__ float s1[256], s2[256];
    s1[threadIdx.x] = lsum; s2[threadIdx.x] = lsq;
    __syncthreads();
    for (int off = 128; off > 0; off >>= 1) {
        if (threadIdx.x < off) {
            s1[threadIdx.x] += s1[threadIdx.x + off];
            s2[threadIdx.x] += s2[threadIdx.x + off];
        }
        __syncthreads();
    }
    const float mean = s1[0] * (1.0f / D_MODEL);
    const float var  = s2[0] * (1.0f / D_MODEL) - mean * mean;
    const float rstd = rsqrtf(var + 1e-3f);
    half_t* orow = out + (long long)row * D_MODEL;
#pragma unroll
    for (int i = 0; i < 3; ++i) {
        int c = threadIdx.x + i * 256;
        orow[c] = (half_t)((v[i] - mean) * rstd * g[c] + b[c]);
    }
}

// ---------------------------------------------------------------------------
// Causal softmax over rows of 1024; scores fp32 -> probs f16
// grid = (1024 rows, 32 batch*head)
// ---------------------------------------------------------------------------
__global__ __launch_bounds__(256) void softmax_k(const float* __restrict__ scores,
                                                 half_t* __restrict__ probs) {
    const int row = blockIdx.x;
    const long long zoff = (long long)blockIdx.y << 20;  // 1024*1024
    const float* sr = scores + zoff + (long long)row * S_LEN;
    half_t* pr = probs + zoff + (long long)row * S_LEN;
    float v[4];
    float lmax = -3.0e38f;
#pragma unroll
    for (int i = 0; i < 4; ++i) {
        int c = threadIdx.x + i * 256;
        v[i] = sr[c] + ((c > row) ? -1e9f : 0.0f);   // mask * -1e9
        lmax = fmaxf(lmax, v[i]);
    }
    __shared__ float sred[256];
    sred[threadIdx.x] = lmax;
    __syncthreads();
    for (int off = 128; off > 0; off >>= 1) {
        if (threadIdx.x < off)
            sred[threadIdx.x] = fmaxf(sred[threadIdx.x], sred[threadIdx.x + off]);
        __syncthreads();
    }
    const float m = sred[0];
    __syncthreads();
    float lsum = 0.f;
#pragma unroll
    for (int i = 0; i < 4; ++i) { v[i] = expf(v[i] - m); lsum += v[i]; }
    sred[threadIdx.x] = lsum;
    __syncthreads();
    for (int off = 128; off > 0; off >>= 1) {
        if (threadIdx.x < off) sred[threadIdx.x] += sred[threadIdx.x + off];
        __syncthreads();
    }
    const float rinv = 1.0f / sred[0];
#pragma unroll
    for (int i = 0; i < 4; ++i) {
        int c = threadIdx.x + i * 256;
        pr[c] = (half_t)(v[i] * rinv);
    }
}

// ---------------------------------------------------------------------------
// WMMA GEMM: C[M,N] = A[M,K] @ B[K,N]  (f16 in, f32 accum).
// Block tile 128x64, 8 waves in 4x2; each wave owns a 32x32 C tile:
// 2 A-frags x 2 B-frags -> 4 v_wmma per 32-k step, fragments reused 2x each.
// Double-buffered LDS (ping-pong), one barrier per k-step, prefetch 2 ahead.
// Batched over blockIdx.z via sA/sB element strides. Epilogue fused via EPI.
// ---------------------------------------------------------------------------
template <int EPI>
__global__ __launch_bounds__(256) void gemm_wmma_k(
    const half_t* __restrict__ A, const half_t* __restrict__ B,
    const float* __restrict__ bias, const float* __restrict__ resid,
    float* __restrict__ outF, half_t* __restrict__ outH,
    half_t* __restrict__ qh, half_t* __restrict__ kth, half_t* __restrict__ vh,
    float* __restrict__ present,
    int K, int ldA, int ldB, long long sA, long long sB) {
    __shared__ half_t As[2][128 * 40];   // [m][k], k-stride 40 halves (16B-aligned rows)
    __shared__ half_t Bs[2][64 * 40];    // transposed: [n][k]

    const int tid = threadIdx.x;
    const int m0 = blockIdx.x * 128;
    const int n0 = blockIdx.y * 64;
    const int z  = blockIdx.z;
    A += (long long)z * sA + (long long)m0 * ldA;
    B += (long long)z * sB + n0;

    const int w = tid >> 5, lane = tid & 31;
    const int wm = w >> 1, wn = w & 1;        // wave tile: rows wm*32, cols wn*32
    const int lhalf = lane >> 4, lmod = lane & 15;
    const int abase = lhalf * 8;              // A frag k-base per ISA layout

    v8f acc[2][2] = {};

    const int arow = tid >> 1, akg = (tid & 1) * 16;  // A: 128 rows x 32 k, 16 halves/thread
    const int brow = tid >> 3, bng = (tid & 7) * 8;   // B: 32 k x 64 n, 8 halves/thread

    const half_t* Aptr = A + (long long)arow * ldA + akg;
    const half_t* Bptr = B + (long long)brow * ldB + bng;

    const int nsteps = K >> 5;

    // prologue: stage tile 0 into buffer 0
    v8h av0 = *(const v8h*)(Aptr);
    v8h av1 = *(const v8h*)(Aptr + 8);
    v8h bv  = *(const v8h*)(Bptr);
    *(v8h*)(&As[0][arow * 40 + akg])     = av0;
    *(v8h*)(&As[0][arow * 40 + akg + 8]) = av1;
#pragma unroll
    for (int q = 0; q < 8; ++q) Bs[0][(bng + q) * 40 + brow] = bv[q];
    __syncthreads();

    for (int i = 0; i < nsteps; ++i) {
        const int cur = i & 1;
        const bool more = (i + 1) < nsteps;
        if (more) {
            const half_t* Ap = Aptr + (i + 1) * 32;
            const half_t* Bp = Bptr + (long long)(i + 1) * 32 * ldB;
            av0 = *(const v8h*)(Ap);
            av1 = *(const v8h*)(Ap + 8);
            bv  = *(const v8h*)(Bp);
            if (i + 2 < nsteps) {
                __builtin_prefetch(Aptr + (i + 2) * 32, 0, 1);
                __builtin_prefetch(Bptr + (long long)(i + 2) * 32 * ldB, 0, 1);
            }
        }

        union Frag { v16h v; v8h h2[2]; } af[2], bf[2];
        // A 16x32 f16 layout: lanes 0-15 hold K 0..7 & 16..23, lanes 16-31 hold 8..15 & 24..31
#pragma unroll
        for (int mi = 0; mi < 2; ++mi) {
            const half_t* Ar = &As[cur][(wm * 32 + mi * 16 + lmod) * 40];
            af[mi].h2[0] = *(const v8h*)(Ar + abase);
            af[mi].h2[1] = *(const v8h*)(Ar + 16 + abase);
        }
        // B 32x16 f16 layout: lanes 0-15 hold K 0..7 & 16..23, lanes 16-31 hold 8..15 & 24..31 (matches A)
#pragma unroll
        for (int ni = 0; ni < 2; ++ni) {
            const half_t* Br = &Bs[cur][(wn * 32 + ni * 16 + lmod) * 40 + abase];
            bf[ni].h2[0] = *(const v8h*)(Br);
            bf[ni].h2[1] = *(const v8h*)(Br + 16);
        }
#pragma unroll
        for (int mi = 0; mi < 2; ++mi)
#pragma unroll
            for (int ni = 0; ni < 2; ++ni)
                acc[mi][ni] = __builtin_amdgcn_wmma_f32_16x16x32_f16(
                    false, af[mi].v, false, bf[ni].v, (short)0, acc[mi][ni], false, false);

        if (more) {
            const int nb = (i + 1) & 1;   // write the buffer nobody is reading
            *(v8h*)(&As[nb][arow * 40 + akg])     = av0;
            *(v8h*)(&As[nb][arow * 40 + akg + 8]) = av1;
#pragma unroll
            for (int q = 0; q < 8; ++q) Bs[nb][(bng + q) * 40 + brow] = bv[q];
        }
        __syncthreads();
    }

    // C/D layout: VGPR r, lane -> m = r + 8*(lane/16), n = lane%16
#pragma unroll
    for (int mi = 0; mi < 2; ++mi)
#pragma unroll
    for (int ni = 0; ni < 2; ++ni)
#pragma unroll
    for (int r = 0; r < 8; ++r) {
        const int gm = m0 + wm * 32 + mi * 16 + 8 * lhalf + r;
        const int gn = n0 + wn * 32 + ni * 16 + lmod;
        float vv = acc[mi][ni][r];
        if constexpr (EPI == EPI_QKV) {
            vv += bias[gn];
            const int b = gm >> 10, s = gm & 1023;
            if (gn < 768) {
                const int h = gn / DHEAD, d = gn % DHEAD;
                qh[(((long long)(b * NHEADS + h)) * S_LEN + s) * DHEAD + d] = (half_t)vv;
            } else if (gn < 1536) {
                const int c = gn - 768, h = c / DHEAD, d = c % DHEAD;
                kth[(((long long)(b * NHEADS + h)) * DHEAD + d) * S_LEN + s] = (half_t)vv;
                present[(((long long)(b * 8 + h)) * S_LEN + s) * DHEAD + d] = vv;        // k
            } else {
                const int c = gn - 1536, h = c / DHEAD, d = c % DHEAD;
                vh[(((long long)(b * NHEADS + h)) * S_LEN + s) * DHEAD + d] = (half_t)vv;
                present[(((long long)(b * 8 + 4 + h)) * S_LEN + s) * DHEAD + d] = vv;    // v
            }
        } else if constexpr (EPI == EPI_SCORE) {
            // faithful bug: multiply by sqrt(d_head)
            outF[((long long)z << 20) + (long long)gm * S_LEN + gn] = vv * 13.856406460551018f;
        } else if constexpr (EPI == EPI_ATTNOUT) {
            const int b = z >> 2, h = z & 3;
            outH[((long long)(b * S_LEN + gm)) * D_MODEL + h * DHEAD + gn] = (half_t)vv;
        } else if constexpr (EPI == EPI_BIASRES) {
            outF[(long long)gm * D_MODEL + gn] =
                vv + bias[gn] + resid[(long long)gm * D_MODEL + gn];
        } else {  // EPI_GELU
            outH[(long long)gm * 3072 + gn] = (half_t)gelu_t(vv + bias[gn]);
        }
    }
}

// ---------------------------------------------------------------------------
extern "C" void kernel_launch(void* const* d_in, const int* in_sizes, int n_in,
                              void* d_out, int out_size, void* d_ws, size_t ws_size,
                              hipStream_t stream) {
    (void)in_sizes; (void)n_in; (void)out_size; (void)ws_size;
    const float* x   = (const float*)d_in[0];
    // d_in[1] = mask (triu) — causality applied analytically in softmax_k
    const float* W1  = (const float*)d_in[2];
    const float* b1  = (const float*)d_in[3];
    const float* W3  = (const float*)d_in[4];
    const float* b3  = (const float*)d_in[5];
    const float* W2  = (const float*)d_in[6];
    const float* b2  = (const float*)d_in[7];
    const float* W4  = (const float*)d_in[8];
    const float* b4  = (const float*)d_in[9];
    const float* g1  = (const float*)d_in[10];
    const float* be1 = (const float*)d_in[11];
    const float* g2  = (const float*)d_in[12];
    const float* be2 = (const float*)d_in[13];

    float* xout    = (float*)d_out;                              // [8,1024,768]
    float* present = (float*)d_out + (long long)ROWS * D_MODEL;  // [8,2,4,1024,192]

    char* ws = (char*)d_ws;
    size_t off = 0;
    auto alloc = [&](size_t bytes) -> void* {
        void* p = ws + off;
        off += (bytes + 255) & ~(size_t)255;
        return p;
    };
    half_t* W1h  = (half_t*)alloc(768LL * 2304 * 2);
    half_t* W3h  = (half_t*)alloc(768LL * 768 * 2);
    half_t* W2h  = (half_t*)alloc(768LL * 3072 * 2);
    half_t* W4h  = (half_t*)alloc(3072LL * 768 * 2);
    half_t* xnh  = (half_t*)alloc((long long)ROWS * 768 * 2);
    half_t* qh   = (half_t*)alloc(32LL * 1024 * 192 * 2);
    half_t* kth  = (half_t*)alloc(32LL * 1024 * 192 * 2);
    half_t* vh   = (half_t*)alloc(32LL * 1024 * 192 * 2);
    float*  sc   = (float*) alloc(32LL * 1024 * 1024 * 4);
    half_t* ph   = (half_t*)alloc(32LL * 1024 * 1024 * 2);
    half_t* ath  = (half_t*)alloc((long long)ROWS * 768 * 2);
    float*  x1   = (float*) alloc((long long)ROWS * 768 * 4);
    half_t* xn2h = (half_t*)alloc((long long)ROWS * 768 * 2);
    half_t* hh   = (half_t*)alloc((long long)ROWS * 3072 * 2);

    int n;
    n = 768 * 2304; f32_to_f16_k<<<(n + 255) / 256, 256, 0, stream>>>(W1, W1h, n);
    n = 768 * 768;  f32_to_f16_k<<<(n + 255) / 256, 256, 0, stream>>>(W3, W3h, n);
    n = 768 * 3072; f32_to_f16_k<<<(n + 255) / 256, 256, 0, stream>>>(W2, W2h, n);
    n = 3072 * 768; f32_to_f16_k<<<(n + 255) / 256, 256, 0, stream>>>(W4, W4h, n);

    // LN1
    layernorm_k<<<ROWS, 256, 0, stream>>>(x, g1, be1, xnh);

    // QKV: [8192,768] @ [768,2304] + b1, split into q/kT/v (f16) and present (f32)
    gemm_wmma_k<EPI_QKV><<<dim3(64, 36, 1), 256, 0, stream>>>(
        xnh, W1h, b1, nullptr, nullptr, nullptr, qh, kth, vh, present,
        768, 768, 2304, 0LL, 0LL);

    // scores[z] = q[z] @ kT[z] * sqrt(192)   (z = b*4+h, M=N=1024, K=192)
    gemm_wmma_k<EPI_SCORE><<<dim3(8, 16, 32), 256, 0, stream>>>(
        qh, kth, nullptr, nullptr, sc, nullptr, nullptr, nullptr, nullptr, nullptr,
        192, 192, 1024, 196608LL, 196608LL);

    // causal softmax
    softmax_k<<<dim3(1024, 32), 256, 0, stream>>>(sc, ph);

    // attn[z] = p[z] @ v[z]  -> concat heads into [b,s,768] f16
    gemm_wmma_k<EPI_ATTNOUT><<<dim3(8, 3, 32), 256, 0, stream>>>(
        ph, vh, nullptr, nullptr, nullptr, ath, nullptr, nullptr, nullptr, nullptr,
        1024, 1024, 192, 1048576LL, 196608LL);

    // x1 = attn @ W3 + b3 + x
    gemm_wmma_k<EPI_BIASRES><<<dim3(64, 12, 1), 256, 0, stream>>>(
        ath, W3h, b3, x, x1, nullptr, nullptr, nullptr, nullptr, nullptr,
        768, 768, 768, 0LL, 0LL);

    // LN2
    layernorm_k<<<ROWS, 256, 0, stream>>>(x1, g2, be2, xn2h);

    // h = gelu(xn2 @ W2 + b2)
    gemm_wmma_k<EPI_GELU><<<dim3(64, 48, 1), 256, 0, stream>>>(
        xn2h, W2h, b2, nullptr, nullptr, hh, nullptr, nullptr, nullptr, nullptr,
        768, 768, 3072, 0LL, 0LL);

    // out = h @ W4 + b4 + x1
    gemm_wmma_k<EPI_BIASRES><<<dim3(64, 12, 1), 256, 0, stream>>>(
        hh, W4h, b4, x1, xout, nullptr, nullptr, nullptr, nullptr, nullptr,
        3072, 3072, 768, 0LL, 0LL);
}
